// QLSTM_23106924052492
// MI455X (gfx1250) — matrix operand order repaired and checked by reading, not compiled, not run on hardware
//
#include <hip/hip_runtime.h>
#include <stdint.h>

// ---------------------------------------------------------------------------
// QLSTM for MI455X (gfx1250): bf16 WMMA (v_wmma_f32_16x16x32_bf16) GEMMs.
// T=128, B=64, F=512, H=1024.
// ---------------------------------------------------------------------------

#define T_ 128
#define B_ 64
#define F_ 512
#define H_ 1024
#define G4 (4 * H_)        // 4096 concatenated gate columns [f|i|o|c]
#define ROWS (T_ * B_)     // 8192
#define NOUT 513
#define NOUTP 528          // padded to 33 n-tiles

typedef __attribute__((ext_vector_type(16))) __bf16 v16bf;
typedef __attribute__((ext_vector_type(8)))  float  v8f;
typedef __attribute__((ext_vector_type(8)))  unsigned short us8;
typedef __attribute__((ext_vector_type(16))) unsigned short us16;
typedef __attribute__((ext_vector_type(4)))  int v4i;

#if __has_builtin(__builtin_amdgcn_global_load_async_to_lds_b128) && \
    __has_builtin(__builtin_amdgcn_s_wait_asynccnt)
#define USE_ASYNC_COPY 1
#else
#define USE_ASYNC_COPY 0
#endif

// ---- workspace layout (bytes) ----
#define OFF_XBF   (size_t)0                       // 8192*512 bf16     = 8 MiB
#define OFF_WSWZ  (size_t)(8388608)               // 512*4096 bf16     = 4 MiB
#define OFF_USWZ  (size_t)(12582912)              // 1024*4096 bf16    = 8 MiB
#define OFF_FSWZ  (size_t)(20971520)              // 1024*528 bf16
#define OFF_G     (size_t)(22052864)              // 8192*4096 bf16    = 64 MiB
#define OFF_HALL  (size_t)(89161728)              // 128*64*1024 bf16  = 16 MiB
#define OFF_C     (size_t)(105938944)             // 64*1024 f32

__device__ __forceinline__ unsigned short f2bf(float f) {
  unsigned int u = __builtin_bit_cast(unsigned int, f);
  u += 0x7FFFu + ((u >> 16) & 1u);            // round-to-nearest-even
  return (unsigned short)(u >> 16);
}
__device__ __forceinline__ float bf2f(unsigned short h) {
  unsigned int u = ((unsigned int)h) << 16;
  return __builtin_bit_cast(float, u);
}
__device__ __forceinline__ float sigm(float x) { return 1.0f / (1.0f + __expf(-x)); }

// Load a 16x32 bf16 A tile (row-major source, stride ldA elements) into the
// contiguous-k order shared with the B tiles: lane half selects K {0..15} vs {16..31}.
__device__ __forceinline__ v16bf load_a_tile(const unsigned short* A, int ldA,
                                             int m0, int k0, int lane) {
  const int lrow  = lane & 15;
  const int khalf = (lane >> 4) << 4;          // 0 or 16
  const unsigned short* ap = A + (size_t)(m0 + lrow) * ldA + k0 + khalf;
  us16 raw = *(const us16*)(ap);               // K = base+0..15
  return __builtin_bit_cast(v16bf, raw);
}

// B tiles are pre-swizzled: tile (kt,nt) stored as 512 bf16, per-lane 16 contiguous.
__device__ __forceinline__ v16bf load_b_tile(const unsigned short* Bs,
                                             int tile_id, int lane) {
  us16 raw = *(const us16*)(Bs + (size_t)tile_id * 512 + lane * 16);
  return __builtin_bit_cast(v16bf, raw);
}

__device__ __forceinline__ v16bf load_b_lds(const unsigned short* Bsh, int g, int lane) {
  us16 raw = *(const us16*)(Bsh + g * 512 + lane * 16);
  return __builtin_bit_cast(v16bf, raw);
}

__device__ __forceinline__ v8f wmma_bf16(v16bf a, v16bf b, v8f c) {
  return __builtin_amdgcn_wmma_f32_16x16x32_bf16(false, a, false, b, (short)0, c,
                                                 false, false);
}

// Hamilton-product component/sign tables: entry (row-block, col-block).
__device__ const int   qcomp[4][4] = {{0,1,2,3},{1,0,3,2},{2,3,0,1},{3,2,1,0}};
__device__ const float qsign[4][4] = {{ 1.f, 1.f, 1.f, 1.f},
                                      {-1.f, 1.f, 1.f,-1.f},
                                      {-1.f,-1.f, 1.f, 1.f},
                                      {-1.f, 1.f,-1.f, 1.f}};

// ---------------------------------------------------------------------------
__global__ void k_convert_x(const float* __restrict__ x, unsigned short* __restrict__ xbf) {
  int i = blockIdx.x * blockDim.x + threadIdx.x;
  if (i < ROWS * F_) xbf[i] = f2bf(x[i]);
}

__global__ void k_zero_c(float* __restrict__ c) {
  int i = blockIdx.x * blockDim.x + threadIdx.x;
  if (i < B_ * H_) c[i] = 0.0f;
}

// Build dense quaternion W (F x 4H) directly in swizzled B-tile order.
__global__ void k_build_W(unsigned short* __restrict__ dst,
                          const float* __restrict__ w0, const float* __restrict__ w1,
                          const float* __restrict__ w2, const float* __restrict__ w3) {
  const int KT = F_ / 32;                                  // 16 k-tiles
  int idx = blockIdx.x * blockDim.x + threadIdx.x;         // = tile*512+lane*16+i
  if (idx >= F_ * G4) return;
  int tile = idx >> 9, lane = (idx >> 4) & 31, i = idx & 15;
  int nt = tile / KT, kt = tile % KT;
  int n = nt * 16 + (lane & 15);
  int k = kt * 32 + ((lane >> 4) << 4) + i;
  int g = n >> 10, j = n & (H_ - 1);
  int rb = k >> 7, p = k & 127;                            // F/4 = 128
  int cb = j >> 8, q = j & 255;                            // H/4 = 256
  const float* w = (g == 0) ? w0 : (g == 1) ? w1 : (g == 2) ? w2 : w3;
  float v = qsign[rb][cb] * w[((size_t)qcomp[rb][cb] * 128 + p) * 256 + q];
  dst[idx] = f2bf(v);
}

// Build dense quaternion U (H x 4H) in swizzled B-tile order.
__global__ void k_build_U(unsigned short* __restrict__ dst,
                          const float* __restrict__ u0, const float* __restrict__ u1,
                          const float* __restrict__ u2, const float* __restrict__ u3) {
  const int KT = H_ / 32;                                  // 32 k-tiles
  int idx = blockIdx.x * blockDim.x + threadIdx.x;
  if (idx >= H_ * G4) return;
  int tile = idx >> 9, lane = (idx >> 4) & 31, i = idx & 15;
  int nt = tile / KT, kt = tile % KT;
  int n = nt * 16 + (lane & 15);
  int k = kt * 32 + ((lane >> 4) << 4) + i;
  int g = n >> 10, j = n & (H_ - 1);
  int rb = k >> 8, p = k & 255;                            // H/4 = 256
  int cb = j >> 8, q = j & 255;
  const float* u = (g == 0) ? u0 : (g == 1) ? u1 : (g == 2) ? u2 : u3;
  float v = qsign[rb][cb] * u[((size_t)qcomp[rb][cb] * 256 + p) * 256 + q];
  dst[idx] = f2bf(v);
}

// Pad fco_w (H x 513) to (H x 528) in swizzled B-tile order.
__global__ void k_build_fco(unsigned short* __restrict__ dst,
                            const float* __restrict__ fw) {
  const int KT = H_ / 32;                                  // 32 k-tiles
  int idx = blockIdx.x * blockDim.x + threadIdx.x;
  if (idx >= H_ * NOUTP) return;
  int tile = idx >> 9, lane = (idx >> 4) & 31, i = idx & 15;
  int nt = tile / KT, kt = tile % KT;
  int n = nt * 16 + (lane & 15);
  int k = kt * 32 + ((lane >> 4) << 4) + i;
  float v = (n < NOUT) ? fw[(size_t)k * NOUT + n] : 0.0f;
  dst[idx] = f2bf(v);
}

// G = Xbf(8192x512) @ Wswz(512x4096) + bias[gate] -> bf16 G(8192x4096)
// Each wave: 1 m-tile x 4 consecutive n-tiles, register double-buffered.
__global__ void k_gemm_in(const unsigned short* __restrict__ Xbf,
                          const unsigned short* __restrict__ Wswz,
                          const float* __restrict__ b0, const float* __restrict__ b1,
                          const float* __restrict__ b2, const float* __restrict__ b3,
                          unsigned short* __restrict__ G) {
  const int KT = F_ / 32;
  int lane = threadIdx.x & 31;
  int wid = blockIdx.x * 4 + (threadIdx.x >> 5);
  int ng = wid & 63, mt = wid >> 6;            // 64 n-groups (4 tiles each), 512 m-tiles
  int nt0 = ng * 4;
  int m0 = mt * 16;
  v8f acc[4] = {{}, {}, {}, {}};
  v16bf a_cur = load_a_tile(Xbf, F_, m0, 0, lane);
  v16bf b_cur[4];
#pragma unroll
  for (int q = 0; q < 4; ++q) b_cur[q] = load_b_tile(Wswz, (nt0 + q) * KT, lane);
  for (int kt = 0; kt < KT; ++kt) {
    int ktn = (kt + 1 < KT) ? kt + 1 : kt;     // clamped: uniform, no divergence
    v16bf a_nxt = load_a_tile(Xbf, F_, m0, ktn * 32, lane);
    v16bf b_nxt[4];
#pragma unroll
    for (int q = 0; q < 4; ++q) b_nxt[q] = load_b_tile(Wswz, (nt0 + q) * KT + ktn, lane);
#pragma unroll
    for (int q = 0; q < 4; ++q) acc[q] = wmma_bf16(a_cur, b_cur[q], acc[q]);
    a_cur = a_nxt;
#pragma unroll
    for (int q = 0; q < 4; ++q) b_cur[q] = b_nxt[q];
  }
#pragma unroll
  for (int q = 0; q < 4; ++q) {
    int n = (nt0 + q) * 16 + (lane & 15);
    int g = n >> 10, j = n & (H_ - 1);
    float bias = (g == 0) ? b0[j] : (g == 1) ? b1[j] : (g == 2) ? b2[j] : b3[j];
#pragma unroll
    for (int r = 0; r < 8; ++r) {
      int m = m0 + r + ((lane >> 4) << 3);
      G[(size_t)m * G4 + n] = f2bf(acc[q][r] + bias);
    }
  }
}

// Stage the 4 gate B-tiles (4KB) for one k-step into LDS.
// 128 threads x 2 chunks of 16B. Async global->LDS when available.
__device__ __forceinline__ void stage_B(unsigned short* dstLds,
                                        const unsigned short* __restrict__ Uswz,
                                        int nt, int kt, int tid) {
#pragma unroll
  for (int c2 = 0; c2 < 2; ++c2) {
    int c = tid * 2 + c2;                      // 0..255
    int g = c >> 6;
    int off = (c & 63) * 8;                    // ushort units (16B chunks)
    const unsigned short* src =
        Uswz + (size_t)((g * 64 + nt) * (H_ / 32) + kt) * 512 + off;
    unsigned short* dst = dstLds + g * 512 + off;
#if USE_ASYNC_COPY
    __builtin_amdgcn_global_load_async_to_lds_b128((v4i*)(void*)src, (v4i*)(void*)dst,
                                                   0, 0);
#else
    *(us8*)dst = *(const us8*)src;
#endif
  }
}

// One recurrent step: acc = h_{t-1} @ [Uf|Ui|Uo|Uc]; fused LSTM cell update.
// Grid: 64 blocks (n-tiles of H), 4 waves/block (batch tiles of 16).
// B tiles double-buffered in LDS (4x less L2 traffic); A double-buffered in regs.
__global__ void k_step(unsigned short* __restrict__ Hall,
                       const unsigned short* __restrict__ G,
                       const unsigned short* __restrict__ Uswz,
                       float* __restrict__ C, int t) {
  const int KT = H_ / 32;
  __shared__ unsigned short Bsh[2][4 * 512];   // 8 KB double buffer
  int tid = threadIdx.x;
  int lane = tid & 31;
  int nt = blockIdx.x;
  int m0 = (tid >> 5) * 16;
  v8f acc[4] = {{}, {}, {}, {}};
  if (t > 0) {
    const unsigned short* hprev = Hall + (size_t)(t - 1) * B_ * H_;
    stage_B(&Bsh[0][0], Uswz, nt, 0, tid);
    v16bf a_cur = load_a_tile(hprev, H_, m0, 0, lane);
    for (int kt = 0; kt < KT; ++kt) {
#if USE_ASYNC_COPY
      __builtin_amdgcn_s_wait_asynccnt(0);
#endif
      __syncthreads();                         // staged buffer kt is ready
      if (kt + 1 < KT) stage_B(&Bsh[(kt + 1) & 1][0], Uswz, nt, kt + 1, tid);
      int ktn = (kt + 1 < KT) ? kt + 1 : kt;
      v16bf a_nxt = load_a_tile(hprev, H_, m0, ktn * 32, lane);
#pragma unroll
      for (int g = 0; g < 4; ++g) {
        v16bf b = load_b_lds(&Bsh[kt & 1][0], g, lane);
        acc[g] = wmma_bf16(a_cur, b, acc[g]);
      }
      a_cur = a_nxt;
    }
  }
  int j = nt * 16 + (lane & 15);
#pragma unroll
  for (int r = 0; r < 8; ++r) {
    int m = m0 + r + ((lane >> 4) << 3);       // batch index
    size_t grow = (size_t)(t * B_ + m) * G4;
    float gf = bf2f(G[grow + j]);
    float gi = bf2f(G[grow + H_ + j]);
    float go = bf2f(G[grow + 2 * H_ + j]);
    float gc = bf2f(G[grow + 3 * H_ + j]);
    float ft = sigm(acc[0][r] + gf);
    float it = sigm(acc[1][r] + gi);
    float ot = sigm(acc[2][r] + go);
    float at = acc[3][r] + gc;
    size_t cidx = (size_t)m * H_ + j;
    float cn = it * tanhf(at) + ft * C[cidx];
    C[cidx] = cn;
    float h = ot * tanhf(cn);
    Hall[(size_t)t * B_ * H_ + cidx] = f2bf(h);
  }
}

// out = Hall(8192x1024) @ Fswz(1024x528) + fco_b  (write only n<513)
// Each wave: 4 m-tiles x 1 n-tile, register double-buffered.
__global__ void k_gemm_out(const unsigned short* __restrict__ Hall,
                           const unsigned short* __restrict__ Fswz,
                           const float* __restrict__ fb, float* __restrict__ out) {
  const int KT = H_ / 32;
  int lane = threadIdx.x & 31;
  int wid = blockIdx.x * 4 + (threadIdx.x >> 5);
  int nt = wid % 33, mg = wid / 33;            // 33 n-tiles, 128 m-groups
  int m0 = mg * 64;
  v8f acc[4] = {{}, {}, {}, {}};
  v16bf b_cur = load_b_tile(Fswz, nt * KT, lane);
  v16bf a_cur[4];
#pragma unroll
  for (int q = 0; q < 4; ++q) a_cur[q] = load_a_tile(Hall, H_, m0 + q * 16, 0, lane);
  for (int kt = 0; kt < KT; ++kt) {
    int ktn = (kt + 1 < KT) ? kt + 1 : kt;
    v16bf b_nxt = load_b_tile(Fswz, nt * KT + ktn, lane);
    v16bf a_nxt[4];
#pragma unroll
    for (int q = 0; q < 4; ++q)
      a_nxt[q] = load_a_tile(Hall, H_, m0 + q * 16, ktn * 32, lane);
#pragma unroll
    for (int q = 0; q < 4; ++q) acc[q] = wmma_bf16(a_cur[q], b_cur, acc[q]);
    b_cur = b_nxt;
#pragma unroll
    for (int q = 0; q < 4; ++q) a_cur[q] = a_nxt[q];
  }
  int n = nt * 16 + (lane & 15);
  if (n >= NOUT) return;
  float bias = fb[n];
#pragma unroll
  for (int q = 0; q < 4; ++q)
#pragma unroll
    for (int r = 0; r < 8; ++r) {
      int m = m0 + q * 16 + r + ((lane >> 4) << 3);
      out[(size_t)m * NOUT + n] = acc[q][r] + bias;
    }
}

// ---------------------------------------------------------------------------
extern "C" void kernel_launch(void* const* d_in, const int* in_sizes, int n_in,
                              void* d_out, int out_size, void* d_ws, size_t ws_size,
                              hipStream_t stream) {
  const float* x     = (const float*)d_in[0];
  const float* wfx_w = (const float*)d_in[1];  const float* wfx_b = (const float*)d_in[2];
  const float* wix_w = (const float*)d_in[3];  const float* wix_b = (const float*)d_in[4];
  const float* wox_w = (const float*)d_in[5];  const float* wox_b = (const float*)d_in[6];
  const float* wcx_w = (const float*)d_in[7];  const float* wcx_b = (const float*)d_in[8];
  const float* ufh_w = (const float*)d_in[9];  const float* uih_w = (const float*)d_in[10];
  const float* uoh_w = (const float*)d_in[11]; const float* uch_w = (const float*)d_in[12];
  const float* fco_w = (const float*)d_in[13]; const float* fco_b = (const float*)d_in[14];

  char* ws = (char*)d_ws;
  unsigned short* Xbf  = (unsigned short*)(ws + OFF_XBF);
  unsigned short* Wswz = (unsigned short*)(ws + OFF_WSWZ);
  unsigned short* Uswz = (unsigned short*)(ws + OFF_USWZ);
  unsigned short* Fswz = (unsigned short*)(ws + OFF_FSWZ);
  unsigned short* G    = (unsigned short*)(ws + OFF_G);
  unsigned short* Hall = (unsigned short*)(ws + OFF_HALL);
  float*          Cbuf = (float*)(ws + OFF_C);

  // preprocess
  k_convert_x<<<(ROWS * F_ + 255) / 256, 256, 0, stream>>>(x, Xbf);
  k_build_W<<<(F_ * G4 + 255) / 256, 256, 0, stream>>>(Wswz, wfx_w, wix_w, wox_w, wcx_w);
  k_build_U<<<(H_ * G4 + 255) / 256, 256, 0, stream>>>(Uswz, ufh_w, uih_w, uoh_w, uch_w);
  k_build_fco<<<(H_ * NOUTP + 255) / 256, 256, 0, stream>>>(Fswz, fco_w);
  k_zero_c<<<(B_ * H_ + 255) / 256, 256, 0, stream>>>(Cbuf);

  // input projections for the whole sequence: 512 m-tiles x 64 n-groups
  k_gemm_in<<<(512 * 64) / 4, 128, 0, stream>>>(Xbf, Wswz, wfx_b, wix_b, wox_b, wcx_b, G);

  // serial recurrence
  for (int t = 0; t < T_; ++t)
    k_step<<<64, 128, 0, stream>>>(Hall, G, Uswz, Cbuf, t);

  // output projection over all timesteps at once: 128 m-groups x 33 n-tiles
  k_gemm_out<<<(128 * 33) / 4, 128, 0, stream>>>(Hall, Fswz, fco_b, (float*)d_out);
}
